// TinySwinUNet_59665685676657
// MI455X (gfx1250) — matrix operand order repaired and checked by reading, not compiled, not run on hardware
//
#include <hip/hip_runtime.h>
#include <hip/hip_bf16.h>

// ---------------------------------------------------------------------------
// TinySwin-UNet forward for gfx1250 (MI455X).  All GEMMs via
// v_wmma_f32_16x16x32_bf16 (f32 -> bf16 converted into LDS, f32 accumulate).
// Memory-bound net: vectorized f32x4 global traffic, branch-free clamped
// addressing (dims aligned: M%16, N%32, K%32), transposed-B LDS tiles so every
// WMMA fragment is a contiguous b128 LDS read, double-buffered GEMM pipeline
// with batched loads (single wait per tile), wave-per-row LayerNorm,
// attention fused entirely in LDS.
// ---------------------------------------------------------------------------

typedef __attribute__((ext_vector_type(16))) __bf16 bf16x16;
typedef __attribute__((ext_vector_type(8)))  __bf16 bf16x8;
typedef __attribute__((ext_vector_type(4)))  __bf16 bf16x4;
typedef __attribute__((ext_vector_type(8)))  float  f32x8;

#define WMMA_BF16(a, b, c) \
  __builtin_amdgcn_wmma_f32_16x16x32_bf16(false, (a), false, (b), (short)0, (c), false, false)

// ----------------------------- WMMA frag loads ------------------------------
// A: 16x32 bf16 (MxK) row-major LDS.  lane<16: M=lane, K={0..15};
// lane>=16: K={16..31}.  One contiguous 32B read.
__device__ inline bf16x16 lds_load_a(const __bf16* base, int lda, int lane) {
  const __bf16* p = base + (lane & 15) * lda + (((lane >> 4) & 1) << 4);
  return *(const bf16x16*)p;
}
// B fragment from n-major tile BsT[n][32]: B(k,n)=BsT[n*32+k].  One 32B read.
__device__ inline bf16x16 lds_load_bt(const __bf16* BsT, int n0, int lane) {
  const __bf16* p = BsT + (size_t)(n0 + (lane & 15)) * 32 + (((lane >> 4) & 1) << 4);
  return *(const bf16x16*)p;
}
// B fragment from n-major tile VsT[n][64] with K offset ks (0/32).
__device__ inline bf16x16 lds_load_bt64(const __bf16* VsT, int n0, int ks, int lane) {
  const __bf16* p = VsT + (size_t)(n0 + (lane & 15)) * 64 + ks + (((lane >> 4) & 1) << 4);
  return *(const bf16x16*)p;
}

__device__ inline float wave_rmax(float v) {
#pragma unroll
  for (int o = 16; o; o >>= 1) v = fmaxf(v, __shfl_xor(v, o, 32));
  return v;
}
__device__ inline float wave_rsum(float v) {
#pragma unroll
  for (int o = 16; o; o >>= 1) v += __shfl_xor(v, o, 32);
  return v;
}

// ------------------------------- generic GEMM -------------------------------
// C[M,N] = A[M,K] @ B[K,N] (+bias) (+GELU | +residual).
// Requires M%16==0, N%32==0, K%32==0 (guaranteed by the net; patch-embed K
// padded 48->64 on the host path).  Block 128x64, 8 waves, wave = 32x32
// (2x2 WMMA frags), K-step 32, double-buffered LDS.
// Load phase and convert/store phase are split so the compiler can issue all
// global loads back-to-back and wait once.
__device__ inline void gemm_load_tiles(const float* __restrict__ A,
                                       const float* __restrict__ Bw,
                                       __bf16* __restrict__ As, __bf16* __restrict__ BsT,
                                       int bm, int bn, int M, int N, int K, int k0, int tid) {
  float4 av[4];
#pragma unroll
  for (int i = 0; i < 4; ++i) {            // A tile 128x32 (1024 float4 groups)
    int g = tid + i * 256;
    int r = g >> 3, kq = (g & 7) << 2;
    int row = min(bm + r, M - 1);          // clamp: OOB rows never stored
    av[i] = *(const float4*)(A + (size_t)row * K + k0 + kq);
  }
  float4 bv[2];
#pragma unroll
  for (int i = 0; i < 2; ++i) {            // B tile 32x64 (512 4-deep k-groups)
    int g = tid + i * 256;
    int n = g & 63, kq = (g >> 6) << 2;
    int col = min(bn + n, N - 1);          // clamp: OOB cols never stored
    const float* bp = Bw + (size_t)(k0 + kq) * N + col;
    bv[i].x = bp[0];
    bv[i].y = bp[(size_t)N];
    bv[i].z = bp[2 * (size_t)N];
    bv[i].w = bp[3 * (size_t)N];
  }
#pragma unroll
  for (int i = 0; i < 4; ++i) {
    int g = tid + i * 256;
    int r = g >> 3, kq = (g & 7) << 2;
    bf16x4 pk = {(__bf16)av[i].x, (__bf16)av[i].y, (__bf16)av[i].z, (__bf16)av[i].w};
    *(bf16x4*)(As + r * 32 + kq) = pk;
  }
#pragma unroll
  for (int i = 0; i < 2; ++i) {            // store transposed [n][k]
    int g = tid + i * 256;
    int n = g & 63, kq = (g >> 6) << 2;
    bf16x4 pk = {(__bf16)bv[i].x, (__bf16)bv[i].y, (__bf16)bv[i].z, (__bf16)bv[i].w};
    *(bf16x4*)(BsT + n * 32 + kq) = pk;
  }
}

__global__ __launch_bounds__(256) void gemm_wmma(
    const float* __restrict__ A, const float* __restrict__ Bw,
    const float* __restrict__ bias, const float* __restrict__ res,
    float* __restrict__ C, int M, int N, int K, int epi) {
  __shared__ __bf16 As[2][128 * 32];
  __shared__ __bf16 BsT[2][64 * 32];
  const int tid = threadIdx.x, lane = tid & 31, wave = tid >> 5;
  const int bm = blockIdx.y * 128, bn = blockIdx.x * 64;
  const int my = (wave >> 1) * 32, nx = (wave & 1) * 32;
  f32x8 acc[2][2] = {};

  gemm_load_tiles(A, Bw, As[0], BsT[0], bm, bn, M, N, K, 0, tid);
  __syncthreads();
  int buf = 0;
  for (int k0 = 0; k0 < K; k0 += 32) {
    if (k0 + 32 < K)                       // fill other buf while computing
      gemm_load_tiles(A, Bw, As[buf ^ 1], BsT[buf ^ 1], bm, bn, M, N, K, k0 + 32, tid);
    bf16x16 af0 = lds_load_a(As[buf] + my * 32, 32, lane);
    bf16x16 af1 = lds_load_a(As[buf] + (my + 16) * 32, 32, lane);
    bf16x16 bf0 = lds_load_bt(BsT[buf], nx, lane);
    bf16x16 bf1 = lds_load_bt(BsT[buf], nx + 16, lane);
    acc[0][0] = WMMA_BF16(af0, bf0, acc[0][0]);
    acc[0][1] = WMMA_BF16(af0, bf1, acc[0][1]);
    acc[1][0] = WMMA_BF16(af1, bf0, acc[1][0]);
    acc[1][1] = WMMA_BF16(af1, bf1, acc[1][1]);
    __syncthreads();
    buf ^= 1;
  }
#pragma unroll
  for (int mi = 0; mi < 2; ++mi) {
    int rbase = bm + my + mi * 16;
    if (rbase >= M) continue;              // uniform (M%16==0)
    int r0 = rbase + ((lane >> 4) << 3);
#pragma unroll
    for (int ni = 0; ni < 2; ++ni) {
      int cbase = bn + nx + ni * 16;
      if (cbase >= N) continue;            // uniform (N%16==0)
      int c = cbase + (lane & 15);
      float bv = bias ? bias[c] : 0.f;
#pragma unroll
      for (int r = 0; r < 8; ++r) {
        int row = r0 + r;
        float v = acc[mi][ni][r] + bv;
        if (epi == 1)      v = 0.5f * v * (1.f + erff(v * 0.70710678118654752f));
        else if (epi == 2) v += res[(size_t)row * N + c];
        C[(size_t)row * N + c] = v;
      }
    }
  }
}

// ------------------------- fused windowed attention -------------------------
// One block per (window, head).  head_dim==32 at every stage.  49 padded to 64.
__device__ inline int swin_region(int u, int H) { return u < H - 7 ? 0 : (u < H - 3 ? 1 : 2); }

__global__ __launch_bounds__(256) void attn_win(
    const float* __restrict__ qkv, const float* __restrict__ relb,
    float* __restrict__ out, int C, int heads, int nWs, int shift, int H) {
  __shared__ __bf16 Qs[64 * 32], Ks2[64 * 32];  // token-major [t][d]
  __shared__ __bf16 VsT[32 * 64];               // dim-major   [d][t]
  __shared__ float Ssm[64 * 64];
  __shared__ __bf16 Ps[64 * 64];
  const int w = blockIdx.x / heads, h = blockIdx.x % heads;
  const int tid = threadIdx.x, lane = tid & 31, wave = tid >> 5;
  const long wtok0 = (long)w * 49;
  const int C3 = 3 * C;
  const int qo = h * 32, ko = C + h * 32, vo = 2 * C + h * 32;
  const float scale = 0.17677669529663687f;  // 32^-0.5

  for (int g = tid; g < 392; g += 256) {       // 49 tokens x 8 float4 groups
    int t = g >> 3, dq = (g & 7) << 2;
    const float* row = qkv + (wtok0 + t) * (long)C3;
    float4 q4 = *(const float4*)(row + qo + dq);
    float4 k4 = *(const float4*)(row + ko + dq);
    float4 v4 = *(const float4*)(row + vo + dq);
    bf16x4 qp = {(__bf16)(q4.x * scale), (__bf16)(q4.y * scale),
                 (__bf16)(q4.z * scale), (__bf16)(q4.w * scale)};
    bf16x4 kp = {(__bf16)k4.x, (__bf16)k4.y, (__bf16)k4.z, (__bf16)k4.w};
    *(bf16x4*)(Qs + t * 32 + dq) = qp;
    *(bf16x4*)(Ks2 + t * 32 + dq) = kp;
    VsT[(dq + 0) * 64 + t] = (__bf16)v4.x;
    VsT[(dq + 1) * 64 + t] = (__bf16)v4.y;
    VsT[(dq + 2) * 64 + t] = (__bf16)v4.z;
    VsT[(dq + 3) * 64 + t] = (__bf16)v4.w;
  }
  for (int e = 1568 + tid; e < 2048; e += 256) {  // zero pad rows 49..63
    Qs[e] = (__bf16)0.f; Ks2[e] = (__bf16)0.f;
  }
  for (int e = tid; e < 480; e += 256) {          // zero pad VsT cols 49..63
    VsT[(e / 15) * 64 + 49 + (e % 15)] = (__bf16)0.f;
  }
  for (int e = tid; e < 512; e += 256) {          // zero P
    bf16x8 z = {};
    *(bf16x8*)(Ps + e * 8) = z;
  }
  __syncthreads();

  // S = Q @ K^T : 4x4 tiles, 2 per wave.  B(k,n) = K[n][k] -> contiguous read.
#pragma unroll
  for (int tt = 0; tt < 2; ++tt) {
    int tile = wave * 2 + tt, mi = tile >> 2, ni = tile & 3;
    bf16x16 af = lds_load_a(Qs + mi * 16 * 32, 32, lane);
    bf16x16 bf = lds_load_bt(Ks2, ni * 16, lane);
    f32x8 acc = {};
    acc = WMMA_BF16(af, bf, acc);
    int r0 = mi * 16 + ((lane >> 4) << 3), c = ni * 16 + (lane & 15);
#pragma unroll
    for (int r = 0; r < 8; ++r) Ssm[(r0 + r) * 64 + c] = acc[r];
  }
  __syncthreads();

  // bias + mask + softmax, write P (bf16)
  const int wim = w % (nWs * nWs);
  const int wh = wim / nWs, ww = wim % nWs;
  for (int r = wave; r < 49; r += 8) {
    int ih = r / 7, iw = r % 7;
    int ri = 0;
    if (shift > 0) ri = swin_region(wh * 7 + ih, H) * 3 + swin_region(ww * 7 + iw, H);
    float v0 = -1e30f, v1 = -1e30f;
    int c0 = lane, c1 = lane + 32;
#pragma unroll
    for (int half = 0; half < 2; ++half) {
      int c = half ? c1 : c0;
      if (c < 49) {
        int jh = c / 7, jw = c % 7;
        int idx = (ih - jh + 6) * 13 + (iw - jw + 6);
        float v = Ssm[r * 64 + c] + relb[idx * heads + h];
        if (shift > 0) {
          int rj = swin_region(wh * 7 + jh, H) * 3 + swin_region(ww * 7 + jw, H);
          if (ri != rj) v -= 100.f;
        }
        if (half) v1 = v; else v0 = v;
      }
    }
    float mx = wave_rmax(fmaxf(v0, v1));
    float e0 = (c0 < 49) ? expf(v0 - mx) : 0.f;
    float e1 = (c1 < 49) ? expf(v1 - mx) : 0.f;
    float inv = 1.f / wave_rsum(e0 + e1);
    Ps[r * 64 + c0] = (__bf16)(e0 * inv);
    Ps[r * 64 + c1] = (__bf16)(e1 * inv);
  }
  __syncthreads();

  // out = P @ V : 4x2 tiles, 1 per wave, K=64 (2 steps), V read n-major.
  {
    int mi = wave >> 1, ni = wave & 1;
    f32x8 acc = {};
#pragma unroll
    for (int ks = 0; ks < 64; ks += 32) {
      bf16x16 af = lds_load_a(Ps + mi * 16 * 64 + ks, 64, lane);
      bf16x16 bf = lds_load_bt64(VsT, ni * 16, ks, lane);
      acc = WMMA_BF16(af, bf, acc);
    }
    int r0 = mi * 16 + ((lane >> 4) << 3), c = ni * 16 + (lane & 15);
#pragma unroll
    for (int r = 0; r < 8; ++r) {
      int m = r0 + r;
      if (m < 49) out[(wtok0 + m) * (long)C + h * 32 + c] = acc[r];
    }
  }
}

// ----------------------------- LayerNorm kernels ----------------------------
// Wave-per-row LN: 8 rows per 256-thread block, shfl reductions, no LDS.
__global__ __launch_bounds__(256) void ln_kernel(
    const float* __restrict__ x, const float* __restrict__ g, const float* __restrict__ b,
    float* __restrict__ y, int C, long T) {
  const long row = blockIdx.x * 8L + (threadIdx.x >> 5);
  if (row >= T) return;
  const int lane = threadIdx.x & 31;
  const float* xr = x + row * (long)C;
  const int C4 = C >> 2;
  float s = 0.f, ss = 0.f;
  for (int q = lane; q < C4; q += 32) {
    float4 v = *(const float4*)(xr + q * 4);
    s += v.x + v.y + v.z + v.w;
    ss += v.x * v.x + v.y * v.y + v.z * v.z + v.w * v.w;
  }
  s = wave_rsum(s); ss = wave_rsum(ss);
  float mean = s / C;
  float rstd = rsqrtf(ss / C - mean * mean + 1e-5f);
  float* yr = y + row * (long)C;
  for (int q = lane; q < C4; q += 32) {
    float4 v = *(const float4*)(xr + q * 4);
    float4 gg = *(const float4*)(g + q * 4);
    float4 bb = *(const float4*)(b + q * 4);
    float4 o4 = {(v.x - mean) * rstd * gg.x + bb.x, (v.y - mean) * rstd * gg.y + bb.y,
                 (v.z - mean) * rstd * gg.z + bb.z, (v.w - mean) * rstd * gg.w + bb.w};
    *(float4*)(yr + q * 4) = o4;
  }
}

// LN fused with roll(-shift) + window partition.  Wave-per-(windowed token).
__global__ __launch_bounds__(256) void ln_winpart(
    const float* __restrict__ x, const float* __restrict__ g, const float* __restrict__ b,
    float* __restrict__ xw, int H, int C, int shift, long T) {
  const long row = blockIdx.x * 8L + (threadIdx.x >> 5);
  if (row >= T) return;
  const int lane = threadIdx.x & 31;
  const int t = (int)(row % 49);
  const long wgi = row / 49;
  const int nWs = H / 7, nW2 = nWs * nWs;
  const int wim = (int)(wgi % nW2), bb_ = (int)(wgi / nW2);
  const int wh = wim / nWs, ww = wim % nWs;
  const int hh = wh * 7 + t / 7, wc = ww * 7 + t % 7;
  const int hs = (hh + shift) % H, wsrc = (wc + shift) % H;
  const float* xr = x + ((long)(bb_ * H + hs) * H + wsrc) * (long)C;
  const int C4 = C >> 2;
  float s = 0.f, ss = 0.f;
  for (int q = lane; q < C4; q += 32) {
    float4 v = *(const float4*)(xr + q * 4);
    s += v.x + v.y + v.z + v.w;
    ss += v.x * v.x + v.y * v.y + v.z * v.z + v.w * v.w;
  }
  s = wave_rsum(s); ss = wave_rsum(ss);
  float mean = s / C;
  float rstd = rsqrtf(ss / C - mean * mean + 1e-5f);
  float* yr = xw + row * (long)C;
  for (int q = lane; q < C4; q += 32) {
    float4 v = *(const float4*)(xr + q * 4);
    float4 gg = *(const float4*)(g + q * 4);
    float4 bb = *(const float4*)(b + q * 4);
    float4 o4 = {(v.x - mean) * rstd * gg.x + bb.x, (v.y - mean) * rstd * gg.y + bb.y,
                 (v.z - mean) * rstd * gg.z + bb.z, (v.w - mean) * rstd * gg.w + bb.w};
    *(float4*)(yr + q * 4) = o4;
  }
}

// ----------------------------- data-movement kernels ------------------------
// im2col for 4x4 patches, K padded 48->64 with zeros.  One float4 per thread.
__global__ void im2col_patch(const float* __restrict__ x, float* __restrict__ A, long total) {
  long id = blockIdx.x * 256L + threadIdx.x;
  if (id >= total) return;                 // total = T0 * 16 groups
  int gq = (int)(id & 15); long m = id >> 4;
  int k = gq * 4;
  float4 v = {0.f, 0.f, 0.f, 0.f};
  if (k < 48) {
    int c = k / 16, pr = (k / 4) % 4;
    int gx = (int)(m % 56); long t = m / 56;
    int gy = (int)(t % 56); int bb_ = (int)(t / 56);
    v = *(const float4*)(x + (((long)bb_ * 3 + c) * 224 + (gy * 4 + pr)) * 224 + gx * 4);
  }
  *(float4*)(A + m * 64 + k) = v;
}

// pad pe_w (48x96) -> (64x96) with zero rows
__global__ void pad_pe_w(const float* __restrict__ w, float* __restrict__ wp) {
  int id = blockIdx.x * 256 + threadIdx.x;
  if (id >= 64 * 96) return;
  int r = id / 96;
  wp[id] = (r < 48) ? w[id] : 0.f;
}

// new_x = shortcut + roll(+shift)(win_rev(proj)) ; float4 over channels
__global__ void winrev_add(const float* __restrict__ shortcut, const float* __restrict__ projw,
                           float* __restrict__ outx, long total4, int H, int C, int shift) {
  long id = blockIdx.x * 256L + threadIdx.x;
  if (id >= total4) return;
  int C4 = C >> 2;
  int cq = (int)(id % C4); long l = id / C4;
  int pos = (int)(l % ((long)H * H)); int bb_ = (int)(l / ((long)H * H));
  int hh = pos / H, wwp = pos % H;
  int hr = (hh - shift + H) % H, wr = (wwp - shift + H) % H;
  int nWs = H / 7;
  long wgi = ((long)bb_ * nWs + hr / 7) * nWs + wr / 7;
  int t = (hr % 7) * 7 + (wr % 7);
  float4 a = *(const float4*)(shortcut + l * (long)C + cq * 4);
  float4 p = *(const float4*)(projw + (wgi * 49 + t) * (long)C + cq * 4);
  float4 o = {a.x + p.x, a.y + p.y, a.z + p.z, a.w + p.w};
  *(float4*)(outx + l * (long)C + cq * 4) = o;
}

// x[B,H,H,C] -> [B,(H/2)^2,4C]  quadrants (0,0),(1,0),(0,1),(1,1); float4
__global__ void merge_gather(const float* __restrict__ x, float* __restrict__ y,
                             long total4, int H, int C) {
  long id = blockIdx.x * 256L + threadIdx.x;
  if (id >= total4) return;
  int C4g = C >> 2, H2 = H / 2;
  int ccq = (int)(id % (4 * C4g)); long l = id / (4 * C4g);
  int hx = (int)(l % H2); long t = l / H2;
  int hy = (int)(t % H2); int bb_ = (int)(t / H2);
  int q = ccq / C4g, cq = ccq % C4g;
  int dh = q & 1, dw = (q >= 2) ? 1 : 0;
  float4 v = *(const float4*)(x + (((long)bb_ * H + (2 * hy + dh)) * H + (2 * hx + dw)) * (long)C + cq * 4);
  *(float4*)(y + id * 4) = v;
}

// y[B,H*H,4C] -> z[B,(2H)^2,C]  (pixel shuffle, ref transpose order); float4
__global__ void expand_rearrange(const float* __restrict__ y, float* __restrict__ z,
                                 long total4, int H, int C) {
  long id = blockIdx.x * 256L + threadIdx.x;
  if (id >= total4) return;
  int C4g = C >> 2;
  int cq = (int)(id % C4g); long l = id / C4g;
  int W2 = 2 * H;
  int wo = (int)(l % W2); long t = l / W2;
  int ho = (int)(t % W2); int bb_ = (int)(t / W2);
  int hh = ho >> 1, i = ho & 1, wwp = wo >> 1, j = wo & 1;
  float4 v = *(const float4*)(y + ((long)bb_ * H * H + (long)hh * H + wwp) * (long)(4 * C) +
                              (i * 2 + j) * C + cq * 4);
  *(float4*)(z + id * 4) = v;
}

__global__ void concat2(const float* __restrict__ a, const float* __restrict__ bsrc,
                        float* __restrict__ o, long total4, int C1, int C2) {
  long id = blockIdx.x * 256L + threadIdx.x;
  if (id >= total4) return;
  int Cc4 = (C1 + C2) >> 2;
  int cq = (int)(id % Cc4); long l = id / Cc4;
  int c = cq * 4;
  float4 v = (c < C1) ? *(const float4*)(a + l * (long)C1 + c)
                      : *(const float4*)(bsrc + l * (long)C2 + (c - C1));
  *(float4*)(o + id * 4) = v;
}

// fused bilinear x4 upsample (align_corners=False) + 1x1 conv (96 -> 2)
__global__ void final_up(const float* __restrict__ feat, const float* __restrict__ fw,
                         const float* __restrict__ fb, float* __restrict__ out) {
  long id = blockIdx.x * 256L + threadIdx.x;
  if (id >= 16L * 224 * 224) return;
  int ow = (int)(id % 224); long t = id / 224;
  int oh = (int)(t % 224); int bb_ = (int)(t / 224);
  float sh = (oh + 0.5f) * 0.25f - 0.5f;
  float sw = (ow + 0.5f) * 0.25f - 0.5f;
  int h0 = (int)floorf(sh), w0 = (int)floorf(sw);
  float fh = sh - h0, fwt = sw - w0;
  int h0c = max(h0, 0), h1c = min(h0 + 1, 55);
  int w0c = max(w0, 0), w1c = min(w0 + 1, 55);
  float w00 = (1.f - fh) * (1.f - fwt), w01 = (1.f - fh) * fwt;
  float w10 = fh * (1.f - fwt), w11 = fh * fwt;
  const float* base = feat + (long)bb_ * 3136 * 96;
  const float* r00 = base + ((long)h0c * 56 + w0c) * 96;
  const float* r01 = base + ((long)h0c * 56 + w1c) * 96;
  const float* r10 = base + ((long)h1c * 56 + w0c) * 96;
  const float* r11 = base + ((long)h1c * 56 + w1c) * 96;
  float a0 = fb[0], a1 = fb[1];
  for (int c = 0; c < 96; c += 4) {
    float4 v00 = *(const float4*)(r00 + c), v01 = *(const float4*)(r01 + c);
    float4 v10 = *(const float4*)(r10 + c), v11 = *(const float4*)(r11 + c);
    float e0 = w00 * v00.x + w01 * v01.x + w10 * v10.x + w11 * v11.x;
    float e1 = w00 * v00.y + w01 * v01.y + w10 * v10.y + w11 * v11.y;
    float e2 = w00 * v00.z + w01 * v01.z + w10 * v10.z + w11 * v11.z;
    float e3 = w00 * v00.w + w01 * v01.w + w10 * v10.w + w11 * v11.w;
    a0 += e0 * fw[c * 2 + 0] + e1 * fw[c * 2 + 2] + e2 * fw[c * 2 + 4] + e3 * fw[c * 2 + 6];
    a1 += e0 * fw[c * 2 + 1] + e1 * fw[c * 2 + 3] + e2 * fw[c * 2 + 5] + e3 * fw[c * 2 + 7];
  }
  out[(((long)bb_ * 2 + 0) * 224 + oh) * 224 + ow] = a0;
  out[(((long)bb_ * 2 + 1) * 224 + oh) * 224 + ow] = a1;
}

// --------------------------------- host side --------------------------------
struct BlockP {
  const float *ln1_g, *ln1_b, *qkv_w, *qkv_b, *rel_bias, *proj_w, *proj_b;
  const float *ln2_g, *ln2_b, *fc1_w, *fc1_b, *fc2_w, *fc2_b;
};

static inline long cdiv(long a, long b) { return (a + b - 1) / b; }

static void launch_gemm(const float* A, const float* Bw, const float* bias, const float* res,
                        float* C, long M, int N, int K, int epi, hipStream_t s) {
  dim3 g((unsigned)cdiv(N, 64), (unsigned)cdiv(M, 128));
  gemm_wmma<<<g, 256, 0, s>>>(A, Bw, bias, res, C, (int)M, N, K, epi);
}

extern "C" void kernel_launch(void* const* d_in, const int* in_sizes, int n_in,
                              void* d_out, int out_size, void* d_ws, size_t ws_size,
                              hipStream_t stream) {
  (void)in_sizes; (void)n_in; (void)out_size; (void)ws_size;
  static const int DIMS_[4] = {96, 192, 384, 768};
  static const int RES_[4] = {56, 28, 14, 7};
  static const int HEADS_[4] = {3, 6, 12, 24};
  const int Bt = 16;

  int idx = 0;
  auto F = [&]() -> const float* { return (const float*)d_in[idx++]; };
  auto RB = [&]() {
    BlockP p;
    p.ln1_g = F(); p.ln1_b = F(); p.qkv_w = F(); p.qkv_b = F(); p.rel_bias = F();
    p.proj_w = F(); p.proj_b = F(); p.ln2_g = F(); p.ln2_b = F();
    p.fc1_w = F(); p.fc1_b = F(); p.fc2_w = F(); p.fc2_b = F();
    return p;
  };

  const float* x = F();
  const float* pe_w = F();    const float* pe_b = F();
  const float* pe_ln_g = F(); const float* pe_ln_b = F();
  BlockP enc[4][2];
  for (int i = 0; i < 4; ++i) for (int j = 0; j < 2; ++j) enc[i][j] = RB();
  const float *mg_g[3], *mg_b[3], *mg_w[3];
  for (int i = 0; i < 3; ++i) { mg_g[i] = F(); mg_b[i] = F(); mg_w[i] = F(); }
  const float *ex_w[3], *ex_g[3], *ex_b[3];
  for (int i = 0; i < 3; ++i) { ex_w[i] = F(); ex_g[i] = F(); ex_b[i] = F(); }
  const float *cc_w[3], *cc_b[3];
  for (int i = 0; i < 3; ++i) { cc_w[i] = F(); cc_b[i] = F(); }
  BlockP dec[3][2];
  for (int i = 0; i < 3; ++i) for (int j = 0; j < 2; ++j) dec[i][j] = RB();
  const float* final_w = F(); const float* final_b = F();

  // workspace carve (floats)
  float* wsf = (float*)d_ws;
  float* TB[2] = {wsf, wsf + 10000000L};
  float* XW   = wsf + 20000000L;   // 10M
  float* QKV  = wsf + 30000000L;   // 15M (also im2col staging)
  float* HID  = wsf + 45000000L;   // 20M
  float* OUTW = wsf + 65000000L;   // 5M  (also padded pe_w staging)
  float* SKIP[3] = {wsf + 70000000L, wsf + 75000000L, wsf + 77500000L};
  float* out = (float*)d_out;

  int ci = 0;

  // ---- patch embed: im2col(K pad 64) -> GEMM(+bias) -> LN ----
  {
    long T0 = (long)Bt * 56 * 56;
    long tot = T0 * 16;                      // float4 groups, K=64
    pad_pe_w<<<(unsigned)cdiv(64 * 96, 256), 256, 0, stream>>>(pe_w, OUTW);
    im2col_patch<<<(unsigned)cdiv(tot, 256), 256, 0, stream>>>(x, QKV, tot);
    launch_gemm(QKV, OUTW, pe_b, nullptr, TB[1], T0, 96, 64, 0, stream);
    ln_kernel<<<(unsigned)cdiv(T0, 8), 256, 0, stream>>>(TB[1], pe_ln_g, pe_ln_b, TB[0], 96, T0);
    ci = 0;
  }

  auto run_block = [&](const BlockP& p, int H, int C, int heads, int shift) {
    long T = (long)Bt * H * H;
    int nWs = H / 7;
    long NW = (long)Bt * nWs * nWs;
    ln_winpart<<<(unsigned)cdiv(NW * 49, 8), 256, 0, stream>>>(
        TB[ci], p.ln1_g, p.ln1_b, XW, H, C, shift, NW * 49);
    launch_gemm(XW, p.qkv_w, p.qkv_b, nullptr, QKV, T, 3 * C, C, 0, stream);
    attn_win<<<(unsigned)(NW * heads), 256, 0, stream>>>(QKV, p.rel_bias, OUTW, C, heads, nWs, shift, H);
    launch_gemm(OUTW, p.proj_w, p.proj_b, nullptr, XW, T, C, C, 0, stream);
    winrev_add<<<(unsigned)cdiv(T * (C >> 2), 256), 256, 0, stream>>>(
        TB[ci], XW, TB[1 - ci], T * (C >> 2), H, C, shift);
    ln_kernel<<<(unsigned)cdiv(T, 8), 256, 0, stream>>>(TB[1 - ci], p.ln2_g, p.ln2_b, XW, C, T);
    launch_gemm(XW, p.fc1_w, p.fc1_b, nullptr, HID, T, 4 * C, C, 1, stream);              // bias+GELU
    launch_gemm(HID, p.fc2_w, p.fc2_b, TB[1 - ci], TB[1 - ci], T, C, 4 * C, 2, stream);   // bias+residual
    ci ^= 1;
  };

  // ---- encoder ----
  for (int li = 0; li < 4; ++li) {
    int H = RES_[li], C = DIMS_[li];
    for (int bi = 0; bi < 2; ++bi)
      run_block(enc[li][bi], H, C, HEADS_[li], (bi % 2 == 0) ? 0 : 3);
    if (li < 3) {
      long T = (long)Bt * H * H;
      hipMemcpyAsync(SKIP[li], TB[ci], (size_t)T * C * sizeof(float),
                     hipMemcpyDeviceToDevice, stream);
      long T2 = (long)Bt * (H / 2) * (H / 2);
      long tot4 = T2 * C;                    // (4C)/4 groups per row
      merge_gather<<<(unsigned)cdiv(tot4, 256), 256, 0, stream>>>(TB[ci], HID, tot4, H, C);
      ln_kernel<<<(unsigned)cdiv(T2, 8), 256, 0, stream>>>(HID, mg_g[li], mg_b[li], XW, 4 * C, T2);
      launch_gemm(XW, mg_w[li], nullptr, nullptr, TB[1 - ci], T2, 2 * C, 4 * C, 0, stream);
      ci ^= 1;
    }
  }

  // ---- decoder ----
  for (int li = 0; li < 3; ++li) {
    int d = 2 - li;
    int C1 = DIMS_[d + 1], H1 = RES_[d + 1];
    long T1 = (long)Bt * H1 * H1;
    launch_gemm(TB[ci], ex_w[li], nullptr, nullptr, HID, T1, 4 * C1, C1, 0, stream);
    int H2 = 2 * H1;
    long T2 = (long)Bt * H2 * H2;
    long tot4 = T2 * (C1 >> 2);
    expand_rearrange<<<(unsigned)cdiv(tot4, 256), 256, 0, stream>>>(HID, XW, tot4, H1, C1);
    ln_kernel<<<(unsigned)cdiv(T2, 8), 256, 0, stream>>>(XW, ex_g[li], ex_b[li], TB[1 - ci], C1, T2);
    int Cs = DIMS_[d];
    long totc4 = T2 * ((C1 + Cs) >> 2);
    concat2<<<(unsigned)cdiv(totc4, 256), 256, 0, stream>>>(TB[1 - ci], SKIP[d], HID, totc4, C1, Cs);
    launch_gemm(HID, cc_w[li], cc_b[li], nullptr, TB[ci], T2, Cs, C1 + Cs, 0, stream);
    for (int bi = 0; bi < 2; ++bi)
      run_block(dec[li][bi], H2, Cs, HEADS_[d], (bi % 2 == 0) ? 0 : 3);
  }

  // ---- final upsample + 1x1 conv ----
  final_up<<<(unsigned)cdiv(16L * 224 * 224, 256), 256, 0, stream>>>(TB[ci], final_w, final_b, out);
}
